// Attention_15573551415521
// MI455X (gfx1250) — matrix operand order repaired and checked by reading, not compiled, not run on hardware
//
#include <hip/hip_runtime.h>

#define DEV __device__ __forceinline__

typedef __attribute__((ext_vector_type(16))) __bf16 v16bf;
typedef __attribute__((ext_vector_type(8))) float v8f;
typedef __attribute__((ext_vector_type(8))) unsigned int v8u;

// bf16 truncation (round-toward-zero): 1 VALU op, or free via d16_hi stores.
DEV unsigned short bftrunc(float f) {
  return (unsigned short)(__builtin_bit_cast(unsigned int, f) >> 16);
}
// Pack two floats' high halves into one dword: single v_perm_b32.
DEV unsigned int bfpack2(float lo, float hi) {
  return __builtin_amdgcn_perm(__builtin_bit_cast(unsigned int, hi),
                               __builtin_bit_cast(unsigned int, lo), 0x07060302u);
}

// A operand (16 x K=32, bf16) from row-major ushort storage (ld even, ushorts).
// ISA layout: lanes 0-15 row m, K 0-7 & 16-23; lanes 16-31 row m, K 8-15 & 24-31.
DEV v16bf load_a_bf16(const unsigned short* p, int ld, int lane) {
  const int m = lane & 15;
  const int kh = (lane >> 4) << 3;           // 0 or 8
  v8u u;
#pragma unroll
  for (int j = 0; j < 4; ++j) {
    u[j]     = *(const unsigned int*)(p + (size_t)m * ld + kh + 2 * j);
    u[j + 4] = *(const unsigned int*)(p + (size_t)m * ld + kh + 16 + 2 * j);
  }
  return __builtin_bit_cast(v16bf, u);
}

// B operand (K=32 x 16) where LDS holds B^T row-major [n][k] (ld even, ushorts).
// ISA layout: lanes 0-15 K 0-7 & 16-23; lanes 16-31 K 8-15 & 24-31, k-pairs per dword.
DEV v16bf load_b_T(const unsigned short* p, int ld, int lane) {
  const int n = lane & 15;
  const int kb = (lane >> 4) << 3;           // 0 or 8
  v8u u;
#pragma unroll
  for (int j = 0; j < 4; ++j) {
    u[j]     = *(const unsigned int*)(p + (size_t)n * ld + kb + 2 * j);
    u[j + 4] = *(const unsigned int*)(p + (size_t)n * ld + kb + 16 + 2 * j);
  }
  return __builtin_bit_cast(v16bf, u);
}

DEV v8f wmma_bf16(v16bf a, v16bf b, v8f c) {
  return __builtin_amdgcn_wmma_f32_16x16x32_bf16(false, a, false, b, (short)0, c,
                                                 false, false);
}

#define XLD 42   // sXT row stride (ushorts) for the GEMM kernels

// ---------------------------------------------------------------------------
// Kernel 1: qkv[b][o][n] (bf16, [c][n]-major) = W_qkv[o][c] @ x[b][c][n]
// grid (N/128, 1536/64, B), block 256.  Each wave: 2x2 16x16 output tiles.
// ---------------------------------------------------------------------------
__global__ __launch_bounds__(256) void k_qkv(const float* __restrict__ x,
                                             const float* __restrict__ w,
                                             unsigned short* __restrict__ qkv,
                                             float qscale) {
  __shared__ unsigned short sW[64 * 32];     // A tiles [o][c]
  __shared__ unsigned short sXT[128 * XLD];  // B^T tiles [n][c]
  const int tid = threadIdx.x, lane = tid & 31, wv = tid >> 5;
  const int N = 4096, C = 512;
  const int n0 = blockIdx.x * 128, o0 = blockIdx.y * 64, b = blockIdx.z;
  const float* xb = x + (size_t)b * C * N;
  const int owp = (wv & 1) * 2;          // o-tile pair
  const int ntp = (wv >> 1) * 2;         // n-tile pair
  v8f acc[2][2] = {};

  for (int c0 = 0; c0 < C; c0 += 32) {
    __syncthreads();
    {  // stage W (64x32): batch loads, pack with v_perm
      float2 wl[4];
#pragma unroll
      for (int t = 0; t < 4; ++t) {
        int e = tid + t * 256, r = e >> 4, cp = (e & 15) * 2;
        wl[t] = *(const float2*)(w + (size_t)(o0 + r) * C + c0 + cp);
      }
#pragma unroll
      for (int t = 0; t < 4; ++t)
        ((unsigned int*)sW)[tid + t * 256] = bfpack2(wl[t].x, wl[t].y);
    }
    {  // stage X^T (transposed): batch loads, truncating d16 stores
      float2 xl[8];
#pragma unroll
      for (int t = 0; t < 8; ++t) {
        int e = tid + t * 256, c = e >> 6, np = (e & 63) * 2;
        xl[t] = *(const float2*)(xb + (size_t)(c0 + c) * N + n0 + np);
      }
#pragma unroll
      for (int t = 0; t < 8; ++t) {
        int e = tid + t * 256, c = e >> 6, np = (e & 63) * 2;
        sXT[np * XLD + c] = bftrunc(xl[t].x);
        sXT[(np + 1) * XLD + c] = bftrunc(xl[t].y);
      }
    }
    __syncthreads();
    v16bf a0 = load_a_bf16(sW + owp * 16 * 32, 32, lane);
    v16bf a1 = load_a_bf16(sW + (owp + 1) * 16 * 32, 32, lane);
    v16bf b0 = load_b_T(sXT + ntp * 16 * XLD, XLD, lane);
    v16bf b1 = load_b_T(sXT + (ntp + 1) * 16 * XLD, XLD, lane);
    acc[0][0] = wmma_bf16(a0, b0, acc[0][0]);
    acc[0][1] = wmma_bf16(a0, b1, acc[0][1]);
    acc[1][0] = wmma_bf16(a1, b0, acc[1][0]);
    acc[1][1] = wmma_bf16(a1, b1, acc[1][1]);
  }
  const float sc = (o0 < 512) ? qscale : 1.0f;
#pragma unroll
  for (int to = 0; to < 2; ++to)
#pragma unroll
    for (int tn = 0; tn < 2; ++tn)
#pragma unroll
      for (int g = 0; g < 8; ++g) {
        int o = o0 + (owp + to) * 16 + g + ((lane >> 4) << 3);
        int nn = n0 + (ntp + tn) * 16 + (lane & 15);
        qkv[((size_t)b * 1536 + o) * N + nn] = bftrunc(acc[to][tn][g] * sc);
      }
}

// ---------------------------------------------------------------------------
// Kernel 2: flash attention. O^T[b][c][n] (bf16) from q,k,v in [c][n] layout.
// grid (N/64, B), block 256, dynamic LDS
// ---------------------------------------------------------------------------
#define QLD 514
#define SLD 65
#define PLD 66
#define SQ_BYTES (64 * QLD * 2)          // 65792  sQ  [query][c]
#define SK_BYTES (64 * QLD * 2)          // 65792  sKT [key][c]
#define SS_BYTES (64 * SLD * 4)          // 16640
#define SP_BYTES (64 * PLD * 2)          // 8448
#define SMEM_K2 (SQ_BYTES + SK_BYTES + SS_BYTES + SP_BYTES + 64 * 4)

__global__ __launch_bounds__(256) void k_attn(const unsigned short* __restrict__ qkv,
                                              unsigned short* __restrict__ ob) {
  extern __shared__ char smem[];
  unsigned short* sQ = (unsigned short*)smem;                          // [64][514]
  unsigned short* sKT = (unsigned short*)(smem + SQ_BYTES);            // [64][514]
  float* sS = (float*)(smem + SQ_BYTES + SK_BYTES);                    // [64][65]
  unsigned short* sP = (unsigned short*)(smem + SQ_BYTES + SK_BYTES + SS_BYTES);
  float* sAl = (float*)(smem + SQ_BYTES + SK_BYTES + SS_BYTES + SP_BYTES);

  const int tid = threadIdx.x, lane = tid & 31, wv = tid >> 5;
  const int N = 4096, C = 512;
  const int i0 = blockIdx.x * 64, b = blockIdx.y;
  const unsigned short* Qp = qkv + (size_t)b * 1536 * N;  // [512][N], pre-scaled
  const unsigned short* Kp = Qp + (size_t)512 * N;
  const unsigned short* Vp = Qp + (size_t)1024 * N;

  // stage Q tile transposed: sQ[i][c]  (batched b64 reads, coalesced along i)
#pragma unroll
  for (int blk = 0; blk < 4; ++blk) {
    uint2 v[8];
#pragma unroll
    for (int t = 0; t < 8; ++t) {
      int e = tid + (blk * 8 + t) * 256, c = e >> 4, ip = (e & 15) * 4;
      v[t] = *(const uint2*)(Qp + (size_t)c * N + i0 + ip);
    }
#pragma unroll
    for (int t = 0; t < 8; ++t) {
      int e = tid + (blk * 8 + t) * 256, c = e >> 4, ip = (e & 15) * 4;
      sQ[ip * QLD + c] = (unsigned short)v[t].x;
      sQ[(ip + 1) * QLD + c] = (unsigned short)(v[t].x >> 16);
      sQ[(ip + 2) * QLD + c] = (unsigned short)v[t].y;
      sQ[(ip + 3) * QLD + c] = (unsigned short)(v[t].y >> 16);
    }
  }

  const int iw2 = (wv & 1) * 2;         // S-phase: 2 query tiles (reuse B)
  const int jt = wv >> 1;               // S-phase: 1 key tile (0..3)
  const int cb = wv * 64;               // PV-phase: 64 channel rows

  v8f acc[4][4] = {};                   // O^T tiles: [ct][it], rows=c cols=query
  float m_i = -1e30f, l_i = 0.0f;

  for (int j0 = 0; j0 < N; j0 += 64) {
    __syncthreads();
    // stage K chunk transposed: sKT[j][c]  (batched b64 reads)
#pragma unroll
    for (int blk = 0; blk < 4; ++blk) {
      uint2 v[8];
#pragma unroll
      for (int t = 0; t < 8; ++t) {
        int e = tid + (blk * 8 + t) * 256, c = e >> 4, jp = (e & 15) * 4;
        v[t] = *(const uint2*)(Kp + (size_t)c * N + j0 + jp);
      }
#pragma unroll
      for (int t = 0; t < 8; ++t) {
        int e = tid + (blk * 8 + t) * 256, c = e >> 4, jp = (e & 15) * 4;
        sKT[jp * QLD + c] = (unsigned short)v[t].x;
        sKT[(jp + 1) * QLD + c] = (unsigned short)(v[t].x >> 16);
        sKT[(jp + 2) * QLD + c] = (unsigned short)v[t].y;
        sKT[(jp + 3) * QLD + c] = (unsigned short)(v[t].y >> 16);
      }
    }
    // prefetch next chunk's K and V rows into cache while this chunk computes
    // (global_prefetch: no counter, fire-and-forget; 1 cacheline covers 64 keys)
    if (j0 + 64 < N) {
      const int c2 = tid * 2;
      __builtin_prefetch(Kp + (size_t)c2 * N + j0 + 64, 0, 1);
      __builtin_prefetch(Kp + (size_t)(c2 + 1) * N + j0 + 64, 0, 1);
      __builtin_prefetch(Vp + (size_t)c2 * N + j0 + 64, 0, 1);
      __builtin_prefetch(Vp + (size_t)(c2 + 1) * N + j0 + 64, 0, 1);
    }
    __syncthreads();
    {   // S = Q^T K : 2 query tiles x 1 key tile, B fragment reused
      v8f s0 = {}, s1 = {};
#pragma unroll
      for (int kc = 0; kc < 16; ++kc) {
        v16bf bb = load_b_T(sKT + jt * 16 * QLD + kc * 32, QLD, lane);
        v16bf a0 = load_a_bf16(sQ + iw2 * 16 * QLD + kc * 32, QLD, lane);
        v16bf a1 = load_a_bf16(sQ + (iw2 + 1) * 16 * QLD + kc * 32, QLD, lane);
        s0 = wmma_bf16(a0, bb, s0);
        s1 = wmma_bf16(a1, bb, s1);
      }
      const int r = (lane >> 4) << 3, cI = lane & 15;
#pragma unroll
      for (int g = 0; g < 8; ++g) {
        sS[(iw2 * 16 + r + g) * SLD + jt * 16 + cI] = s0[g];
        sS[((iw2 + 1) * 16 + r + g) * SLD + jt * 16 + cI] = s1[g];
      }
    }
    __syncthreads();
    if (tid < 64) {                     // online softmax, 1 thread / query row
      float mx = m_i;
#pragma unroll 8
      for (int j = 0; j < 64; ++j) mx = fmaxf(mx, sS[tid * SLD + j]);
      float alpha = __expf(m_i - mx);
      float sum = 0.f;
#pragma unroll 4
      for (int jp = 0; jp < 64; jp += 2) {   // packed P stores (v_perm + b32)
        float p0 = __expf(sS[tid * SLD + jp] - mx);
        float p1 = __expf(sS[tid * SLD + jp + 1] - mx);
        sum += p0 + p1;
        ((unsigned int*)sP)[(tid * PLD + jp) >> 1] = bfpack2(p0, p1);
      }
      m_i = mx;
      l_i = l_i * alpha + sum;
      sAl[tid] = alpha;
    }
    __syncthreads();
#pragma unroll
    for (int it = 0; it < 4; ++it) {    // rescale: alpha indexed by query col
      float al = sAl[it * 16 + (lane & 15)];
#pragma unroll
      for (int ct = 0; ct < 4; ++ct) acc[ct][it] *= al;
    }
#pragma unroll
    for (int js = 0; js < 2; ++js) {    // O^T += V * P^T  (V streamed from L2)
      v16bf bt[4];
#pragma unroll
      for (int it = 0; it < 4; ++it)
        bt[it] = load_b_T(sP + it * 16 * PLD + js * 32, PLD, lane);
#pragma unroll
      for (int ct = 0; ct < 4; ++ct) {
        const unsigned short* vp = Vp + (size_t)(cb + ct * 16) * N + j0 + js * 32;
        v16bf a = load_a_bf16(vp, N, lane);
#pragma unroll
        for (int it = 0; it < 4; ++it) acc[ct][it] = wmma_bf16(a, bt[it], acc[ct][it]);
      }
    }
  }
  __syncthreads();
  if (tid < 64) sAl[tid] = 1.0f / l_i;
  __syncthreads();
  unsigned short* outp = ob + (size_t)b * C * N;
#pragma unroll
  for (int it = 0; it < 4; ++it) {
    float rl = sAl[it * 16 + (lane & 15)];
#pragma unroll
    for (int ct = 0; ct < 4; ++ct)
#pragma unroll
      for (int g = 0; g < 8; ++g) {
        int c = cb + ct * 16 + g + ((lane >> 4) << 3);
        int i = i0 + it * 16 + (lane & 15);
        outp[(size_t)c * N + i] = bftrunc(acc[ct][it][g] * rl);
      }
  }
}

// ---------------------------------------------------------------------------
// Kernel 3: out[b][o][n] (fp32) = W_out[o][c] @ O^T[b][c][n]
// grid (N/128, 512/64, B), block 256.  Each wave: 2x2 16x16 output tiles.
// ---------------------------------------------------------------------------
__global__ __launch_bounds__(256) void k_proj(const unsigned short* __restrict__ ob,
                                              const float* __restrict__ w,
                                              float* __restrict__ out) {
  __shared__ unsigned short sW[64 * 32];
  __shared__ unsigned short sXT[128 * XLD];
  const int tid = threadIdx.x, lane = tid & 31, wv = tid >> 5;
  const int N = 4096, C = 512;
  const int n0 = blockIdx.x * 128, o0 = blockIdx.y * 64, b = blockIdx.z;
  const unsigned short* xb = ob + (size_t)b * C * N;
  const int owp = (wv & 1) * 2, ntp = (wv >> 1) * 2;
  v8f acc[2][2] = {};

  for (int c0 = 0; c0 < C; c0 += 32) {
    __syncthreads();
    {
      float2 wl[4];
#pragma unroll
      for (int t = 0; t < 4; ++t) {
        int e = tid + t * 256, r = e >> 4, cp = (e & 15) * 2;
        wl[t] = *(const float2*)(w + (size_t)(o0 + r) * C + c0 + cp);
      }
#pragma unroll
      for (int t = 0; t < 4; ++t)
        ((unsigned int*)sW)[tid + t * 256] = bfpack2(wl[t].x, wl[t].y);
    }
    {  // bf16 source: batched b32 reads
      unsigned int xl[8];
#pragma unroll
      for (int t = 0; t < 8; ++t) {
        int e = tid + t * 256, c = e >> 6, np = (e & 63) * 2;
        xl[t] = *(const unsigned int*)(xb + (size_t)(c0 + c) * N + n0 + np);
      }
#pragma unroll
      for (int t = 0; t < 8; ++t) {
        int e = tid + t * 256, c = e >> 6, np = (e & 63) * 2;
        sXT[np * XLD + c] = (unsigned short)xl[t];
        sXT[(np + 1) * XLD + c] = (unsigned short)(xl[t] >> 16);
      }
    }
    __syncthreads();
    v16bf a0 = load_a_bf16(sW + owp * 16 * 32, 32, lane);
    v16bf a1 = load_a_bf16(sW + (owp + 1) * 16 * 32, 32, lane);
    v16bf b0 = load_b_T(sXT + ntp * 16 * XLD, XLD, lane);
    v16bf b1 = load_b_T(sXT + (ntp + 1) * 16 * XLD, XLD, lane);
    acc[0][0] = wmma_bf16(a0, b0, acc[0][0]);
    acc[0][1] = wmma_bf16(a0, b1, acc[0][1]);
    acc[1][0] = wmma_bf16(a1, b0, acc[1][0]);
    acc[1][1] = wmma_bf16(a1, b1, acc[1][1]);
  }
#pragma unroll
  for (int to = 0; to < 2; ++to)
#pragma unroll
    for (int tn = 0; tn < 2; ++tn)
#pragma unroll
      for (int g = 0; g < 8; ++g) {
        int o = o0 + (owp + to) * 16 + g + ((lane >> 4) << 3);
        int nn = n0 + (ntp + tn) * 16 + (lane & 15);
        out[((size_t)b * C + o) * N + nn] = acc[to][tn][g];
      }
}

extern "C" void kernel_launch(void* const* d_in, const int* in_sizes, int n_in,
                              void* d_out, int out_size, void* d_ws, size_t ws_size,
                              hipStream_t stream) {
  (void)in_sizes; (void)n_in; (void)out_size; (void)ws_size;
  const float* x = (const float*)d_in[0];
  const float* wqkv = (const float*)d_in[1];
  const float* wout = (const float*)d_in[2];
  float* out = (float*)d_out;

  unsigned short* qkv = (unsigned short*)d_ws;               // 8*1536*4096 bf16
  unsigned short* ob = qkv + (size_t)8 * 1536 * 4096;        // 8*512*4096 bf16

  hipFuncSetAttribute((const void*)k_attn,
                      hipFuncAttributeMaxDynamicSharedMemorySize, SMEM_K2);

  const float qscale = 0.044194173824159216f;                // 512^-0.5
  dim3 blk(256);
  k_qkv<<<dim3(32, 24, 8), blk, 0, stream>>>(x, wqkv, qkv, qscale);
  k_attn<<<dim3(64, 8), blk, SMEM_K2, stream>>>(qkv, ob);
  k_proj<<<dim3(32, 8, 8), blk, 0, stream>>>(ob, wout, out);
}
